// Encoder_48619029790896
// MI455X (gfx1250) — matrix operand order repaired and checked by reading, not compiled, not run on hardware
//
#include <hip/hip_runtime.h>

// ---------------------------------------------------------------------------
// Types for CDNA5 WMMA / TDM (wave32)
// ---------------------------------------------------------------------------
typedef __attribute__((ext_vector_type(16))) __bf16   v16bf;
typedef __attribute__((ext_vector_type(8)))  __bf16   v8bf;
typedef __attribute__((ext_vector_type(8)))  float    v8f;
typedef __attribute__((ext_vector_type(4)))  unsigned u32x4;
typedef __attribute__((ext_vector_type(8)))  int      i32x8;
typedef __attribute__((ext_vector_type(4)))  int      i32x4;

#define NN       20000
#define EE       160000
#define IN_DIM   128
#define EDGE_DIM 32
#define HID      128
#define OUTD     64
#define MAX_K    512     // largest GEMM K; bounds the LDS A-tile

// ---------------------------------------------------------------------------
// Helpers: ordered-uint encoding of floats for atomic max
// ---------------------------------------------------------------------------
__device__ __forceinline__ unsigned fflip(float f) {
    unsigned u = __float_as_uint(f);
    return (u & 0x80000000u) ? ~u : (u | 0x80000000u);
}
__device__ __forceinline__ float funflip(unsigned u) {
    return (u & 0x80000000u) ? __uint_as_float(u & 0x7FFFFFFFu)
                             : __uint_as_float(~u);
}

// ---------------------------------------------------------------------------
// Utility kernels
// ---------------------------------------------------------------------------
__global__ void cvt_f32_bf16_kernel(const float* __restrict__ src,
                                    __bf16* __restrict__ dst, int n) {
    int i = blockIdx.x * blockDim.x + threadIdx.x;
    if (i < n) dst[i] = (__bf16)src[i];
}

__global__ void copy_f32_kernel(const float* __restrict__ src,
                                float* __restrict__ dst, int n) {
    int i = blockIdx.x * blockDim.x + threadIdx.x;
    if (i < n) dst[i] = src[i];
}

__global__ void zero_u32_kernel(unsigned* __restrict__ p, int n) {
    int i = blockIdx.x * blockDim.x + threadIdx.x;
    if (i < n) p[i] = 0u;
}

// ---------------------------------------------------------------------------
// Tiled bf16 WMMA GEMM:  C[M,Nn] = A[M,K] @ W[Nn,K]^T (+ bias[Nn])
//
// Grid: (ceil(tilesN/8), tilesM). A block owns one 16-row A tile (uniform tm);
// wave 0 DMAs it into LDS with the Tensor Data Mover (2D D#: 16 x K bf16,
// stride K), waits on TENSORcnt, then each wave computes a 16x64 tile reading
// A fragments from LDS (ds_load_b128) and B fragments from global (weights
// are L2-resident). A and B both use contiguous k order (k = 16*half + s).
// ---------------------------------------------------------------------------
__global__ void wmma_gemm_bf16_kernel(const __bf16* __restrict__ A,
                                      const __bf16* __restrict__ W,
                                      const float* __restrict__ bias,
                                      float* __restrict__ C,
                                      int M, int Nn, int K) {
    __shared__ __bf16 shA[16 * MAX_K];

    int w    = threadIdx.x >> 5;
    int lane = threadIdx.x & 31;
    int tilesN = Nn >> 6;                 // 64 cols per wave
    int tm = blockIdx.y;                  // block-uniform row tile
    int tn = blockIdx.x * 8 + w;

    // ---- stage A tile (16 rows x K) into LDS ---------------------------
    if (w == 0) {
#if __has_builtin(__builtin_amdgcn_tensor_load_to_lds) && \
    __has_builtin(__builtin_amdgcn_s_wait_tensorcnt)
        unsigned lds = (unsigned)(unsigned long long)(const void*)shA;
        unsigned long long ga =
            (unsigned long long)(const void*)(A + (size_t)tm * 16 * K);

        // D# group 0: count=1 | lds_addr | global_addr | type=2
        u32x4 g0 = { 1u,
                     lds,
                     (unsigned)(ga & 0xFFFFFFFFull),
                     (unsigned)((ga >> 32) & 0x01FFFFFFull) | (2u << 30) };
        // D# group 1: data_size=2B; tensor_dim0=K, tensor_dim1=M;
        //             tile_dim0=K, tile_dim1=16; tensor_dim0_stride=K
        i32x8 g1 = { (int)(1u << 16),
                     (int)((unsigned)(K & 0xFFFF) << 16),
                     (int)(((unsigned)K >> 16) | ((unsigned)(M & 0xFFFF) << 16)),
                     (int)((((unsigned)M >> 16) & 0xFFFFu) |
                           ((unsigned)(K & 0xFFFF) << 16)),
                     16,                     // tile_dim1 = 16 rows
                     K,                      // tensor_dim0_stride[31:0]
                     0, 0 };
        i32x4 g2 = { 0, 0, 0, 0 };
        i32x4 g3 = { 0, 0, 0, 0 };
#if defined(__clang_major__) && (__clang_major__ >= 23)
        i32x8 g4 = { 0, 0, 0, 0, 0, 0, 0, 0 };
        __builtin_amdgcn_tensor_load_to_lds(g0, g1, g2, g3, g4, 0);
#else
        __builtin_amdgcn_tensor_load_to_lds(g0, g1, g2, g3, 0);
#endif
        __builtin_amdgcn_s_wait_tensorcnt(0);
#else
        // Fallback: wave-0 cooperative copy
        const __bf16* srcA = A + (size_t)tm * 16 * K;
        for (int i = lane; i < 16 * K; i += 32) shA[i] = srcA[i];
#endif
    }
    __syncthreads();

    if (tn >= tilesN) return;

    int half = lane >> 4;                 // lane half selects K sub-range
    int l16  = lane & 15;

    v8f acc0 = {}, acc1 = {}, acc2 = {}, acc3 = {};

    // A fragment (from LDS): row = l16, contiguous chunk [kbase,+16)
    const __bf16* arow = shA + (size_t)l16 * K;
    const int kbase = half * 16;

    // B fragments: column n = tile + l16; contiguous K half (16 elems) of W row
    const __bf16* w0 = W + (size_t)(tn * 64 +  0 + l16) * K + half * 16;
    const __bf16* w1 = W + (size_t)(tn * 64 + 16 + l16) * K + half * 16;
    const __bf16* w2 = W + (size_t)(tn * 64 + 32 + l16) * K + half * 16;
    const __bf16* w3 = W + (size_t)(tn * 64 + 48 + l16) * K + half * 16;

    for (int kt = 0; kt < K; kt += 32) {
        v8bf alo = *(const v8bf*)(arow + kt + kbase);
        v8bf ahi = *(const v8bf*)(arow + kt + kbase + 8);
        v16bf a;
#pragma unroll
        for (int i = 0; i < 8; ++i) { a[i] = alo[i]; a[i + 8] = ahi[i]; }

        v16bf b0 = *(const v16bf*)(w0 + kt);
        v16bf b1 = *(const v16bf*)(w1 + kt);
        v16bf b2 = *(const v16bf*)(w2 + kt);
        v16bf b3 = *(const v16bf*)(w3 + kt);

        acc0 = __builtin_amdgcn_wmma_f32_16x16x32_bf16(false, a, false, b0,
                                                       (short)0, acc0, false, false);
        acc1 = __builtin_amdgcn_wmma_f32_16x16x32_bf16(false, a, false, b1,
                                                       (short)0, acc1, false, false);
        acc2 = __builtin_amdgcn_wmma_f32_16x16x32_bf16(false, a, false, b2,
                                                       (short)0, acc2, false, false);
        acc3 = __builtin_amdgcn_wmma_f32_16x16x32_bf16(false, a, false, b3,
                                                       (short)0, acc3, false, false);
    }

    float bn0 = 0.f, bn1 = 0.f, bn2 = 0.f, bn3 = 0.f;
    if (bias) {
        bn0 = bias[tn * 64 +  0 + l16];
        bn1 = bias[tn * 64 + 16 + l16];
        bn2 = bias[tn * 64 + 32 + l16];
        bn3 = bias[tn * 64 + 48 + l16];
    }

    // D layout: lane column = l16, rows r + 8*half
    int rowbase = tm * 16 + 8 * half;
    size_t colbase = (size_t)tn * 64 + l16;
#pragma unroll
    for (int r = 0; r < 8; ++r) {
        size_t ro = (size_t)(rowbase + r) * Nn + colbase;
        C[ro +  0] = acc0[r] + bn0;
        C[ro + 16] = acc1[r] + bn1;
        C[ro + 32] = acc2[r] + bn2;
        C[ro + 48] = acc3[r] + bn3;
    }
}

// ---------------------------------------------------------------------------
// Attention: one wave per (edge, head).
//   alpha = dot(q[dst,h,:], k[src,h,:] + e[edge,h,:]) * scale
//   segment max via atomicMax on ordered uint.
//   Lane 0 also prefetches v[src,h,:] (global_prefetch_b8) to warm the 192MB
//   L2 for the scatter pass that gathers the same rows.
// ---------------------------------------------------------------------------
__global__ void attn_alpha_kernel(const float* __restrict__ qkvs, int ld,
                                  int koff, int voff,
                                  const float* __restrict__ ef,
                                  const long long* __restrict__ ei,
                                  int E, int H, int C, float scale,
                                  float* __restrict__ alpha,
                                  unsigned* __restrict__ amax) {
    int wid  = (blockIdx.x * blockDim.x + threadIdx.x) >> 5;
    int lane = threadIdx.x & 31;
    if (wid >= E * H) return;
    int e = wid / H;
    int h = wid - e * H;
    int src = (int)ei[e];
    int dst = (int)ei[E + e];

    const float* q  = qkvs + (size_t)dst * ld + h * C;          // qoff == 0
    const float* k  = qkvs + (size_t)src * ld + koff + h * C;
    const float* er = ef   + (size_t)e * (H * C) + h * C;

    float s = 0.f;
    for (int c = lane; c < C; c += 32) s += q[c] * (k[c] + er[c]);
#pragma unroll
    for (int o = 16; o; o >>= 1) s += __shfl_xor(s, o, 32);
    s *= scale;
    if (lane == 0) {
        alpha[(size_t)e * H + h] = s;
        atomicMax(&amax[(size_t)dst * H + h], fflip(s));
        // warm L2 for the scatter pass's v[src] gather
        __builtin_prefetch(qkvs + (size_t)src * ld + voff + h * C, 0, 1);
    }
}

// exp(alpha - amax[dst]) ; denom += ex  (one thread per edge-head)
__global__ void attn_exp_kernel(float* __restrict__ alpha,
                                const long long* __restrict__ ei,
                                int E, int H,
                                const unsigned* __restrict__ amax,
                                float* __restrict__ denom) {
    int i = blockIdx.x * blockDim.x + threadIdx.x;
    if (i >= E * H) return;
    int e = i / H;
    int h = i - e * H;
    int dst = (int)ei[E + e];
    float m = funflip(amax[(size_t)dst * H + h]);
    float v = __expf(alpha[i] - m);
    alpha[i] = v;
    atomicAdd(&denom[(size_t)dst * H + h], v);
}

// out[dst,h,:] += (v[src,h,:] + e[edge,h,:]) * (ex / denom[dst,h])
__global__ void attn_scatter_kernel(const float* __restrict__ qkvs, int ld,
                                    int voff,
                                    const float* __restrict__ ef,
                                    const long long* __restrict__ ei,
                                    int E, int H, int C,
                                    const float* __restrict__ alpha,
                                    const float* __restrict__ denom,
                                    float* __restrict__ out) {
    int wid  = (blockIdx.x * blockDim.x + threadIdx.x) >> 5;
    int lane = threadIdx.x & 31;
    if (wid >= E * H) return;
    int e = wid / H;
    int h = wid - e * H;
    int src = (int)ei[e];
    int dst = (int)ei[E + e];

    float w = alpha[(size_t)e * H + h] / denom[(size_t)dst * H + h];
    const float* v  = qkvs + (size_t)src * ld + voff + h * C;
    const float* er = ef   + (size_t)e * (H * C) + h * C;
    float* o = out + (size_t)dst * (H * C) + h * C;
    for (int c = lane; c < C; c += 32)
        atomicAdd(&o[c], (v[c] + er[c]) * w);
}

// h = relu(agg + skip) -> bf16   (layer 1, dim 512, skip at col 1536, ld 2048)
__global__ void finalize1_kernel(const float* __restrict__ agg,
                                 const float* __restrict__ qkvs1,
                                 __bf16* __restrict__ hb, int n) {
    int i = blockIdx.x * blockDim.x + threadIdx.x;
    if (i >= n) return;
    int row = i >> 9;           // /512
    int j   = i & 511;
    float v = agg[i] + qkvs1[(size_t)row * 2048 + 1536 + j];
    hb[i] = (__bf16)(v > 0.f ? v : 0.f);
}

// z = agg + skip   (layer 2, dim 64, skip at col 192, ld 256)
__global__ void finalize2_kernel(const float* __restrict__ agg,
                                 const float* __restrict__ qkvs2,
                                 float* __restrict__ z, int n) {
    int i = blockIdx.x * blockDim.x + threadIdx.x;
    if (i >= n) return;
    int row = i >> 6;           // /64
    int j   = i & 63;
    z[i] = agg[i] + qkvs2[(size_t)row * 256 + 192 + j];
}

// ---------------------------------------------------------------------------
// Host launcher
// ---------------------------------------------------------------------------
extern "C" void kernel_launch(void* const* d_in, const int* in_sizes, int n_in,
                              void* d_out, int out_size, void* d_ws, size_t ws_size,
                              hipStream_t stream) {
    const float*     x   = (const float*)d_in[0];
    const long long* ei  = (const long long*)d_in[1];
    const float*     ea  = (const float*)d_in[2];
    const float*     Wq1 = (const float*)d_in[3];
    const float*     bq1 = (const float*)d_in[4];
    const float*     Wk1 = (const float*)d_in[5];
    const float*     bk1 = (const float*)d_in[6];
    const float*     Wv1 = (const float*)d_in[7];
    const float*     bv1 = (const float*)d_in[8];
    const float*     We1 = (const float*)d_in[9];
    const float*     Ws1 = (const float*)d_in[10];
    const float*     bs1 = (const float*)d_in[11];
    const float*     Wq2 = (const float*)d_in[12];
    const float*     bq2 = (const float*)d_in[13];
    const float*     Wk2 = (const float*)d_in[14];
    const float*     bk2 = (const float*)d_in[15];
    const float*     Wv2 = (const float*)d_in[16];
    const float*     bv2 = (const float*)d_in[17];
    const float*     We2 = (const float*)d_in[18];
    const float*     Ws2 = (const float*)d_in[19];
    const float*     bs2 = (const float*)d_in[20];

    char*  ws  = (char*)d_ws;
    size_t off = 0;
    auto alloc = [&](size_t bytes) -> char* {
        char* p = ws + off;
        off = (off + bytes + 255) & ~(size_t)255;
        return p;
    };

    __bf16*   xb     = (__bf16*)alloc((size_t)NN * IN_DIM * 2);
    __bf16*   eab    = (__bf16*)alloc((size_t)EE * EDGE_DIM * 2);
    __bf16*   W1     = (__bf16*)alloc((size_t)2048 * 128 * 2);   // [Wq1;Wk1;Wv1;Ws1]
    float*    b1     = (float*) alloc(2048 * 4);
    __bf16*   We1b   = (__bf16*)alloc((size_t)512 * 32 * 2);
    __bf16*   W2     = (__bf16*)alloc((size_t)256 * 512 * 2);    // [Wq2;Wk2;Wv2;Ws2]
    float*    b2     = (float*) alloc(256 * 4);
    __bf16*   We2b   = (__bf16*)alloc((size_t)64 * 32 * 2);
    float*    qkvs1  = (float*) alloc((size_t)NN * 2048 * 4);
    float*    e1     = (float*) alloc((size_t)EE * 512 * 4);
    float*    alpha1 = (float*) alloc((size_t)EE * 4 * 4);
    unsigned* amax1  = (unsigned*)alloc((size_t)NN * 4 * 4);
    float*    den1   = (float*) alloc((size_t)NN * 4 * 4);
    float*    out1   = (float*) alloc((size_t)NN * 512 * 4);
    __bf16*   hb     = (__bf16*)alloc((size_t)NN * 512 * 2);
    float*    qkvs2  = (float*) alloc((size_t)NN * 256 * 4);
    float*    e2     = (float*) alloc((size_t)EE * 64 * 4);
    float*    alpha2 = (float*) alloc((size_t)EE * 4);
    unsigned* amax2  = (unsigned*)alloc((size_t)NN * 4);
    float*    den2   = (float*) alloc((size_t)NN * 4);
    float*    out2   = (float*) alloc((size_t)NN * 64 * 4);

    auto cdiv = [](long long a, long long b) { return (int)((a + b - 1) / b); };

#define CVT(src, dst, n)  cvt_f32_bf16_kernel<<<cdiv((n), 256), 256, 0, stream>>>((src), (dst), (n))
#define CPY(src, dst, n)  copy_f32_kernel<<<cdiv((n), 256), 256, 0, stream>>>((src), (dst), (n))
#define ZERO(p, n)        zero_u32_kernel<<<cdiv((n), 256), 256, 0, stream>>>((unsigned*)(p), (n))

    // --- conversions / weight packing -------------------------------------
    CVT(x,  xb,  NN * IN_DIM);
    CVT(ea, eab, EE * EDGE_DIM);
    CVT(Wq1, W1 + (size_t)0 * 512 * 128, 512 * 128);
    CVT(Wk1, W1 + (size_t)1 * 512 * 128, 512 * 128);
    CVT(Wv1, W1 + (size_t)2 * 512 * 128, 512 * 128);
    CVT(Ws1, W1 + (size_t)3 * 512 * 128, 512 * 128);
    CPY(bq1, b1 + 0,    512); CPY(bk1, b1 + 512,  512);
    CPY(bv1, b1 + 1024, 512); CPY(bs1, b1 + 1536, 512);
    CVT(We1, We1b, 512 * 32);
    CVT(Wq2, W2 + (size_t)0 * 64 * 512, 64 * 512);
    CVT(Wk2, W2 + (size_t)1 * 64 * 512, 64 * 512);
    CVT(Wv2, W2 + (size_t)2 * 64 * 512, 64 * 512);
    CVT(Ws2, W2 + (size_t)3 * 64 * 512, 64 * 512);
    CPY(bq2, b2 + 0,   64); CPY(bk2, b2 + 64,  64);
    CPY(bv2, b2 + 128, 64); CPY(bs2, b2 + 192, 64);
    CVT(We2, We2b, 64 * 32);

    auto gemm = [&](const __bf16* A, const __bf16* W, const float* bias,
                    float* C, int M, int Nn, int K) {
        int tilesM = M >> 4;
        int tilesN = Nn >> 6;
        dim3 grid((unsigned)cdiv(tilesN, 8), (unsigned)tilesM);
        wmma_gemm_bf16_kernel<<<grid, 256, 0, stream>>>(A, W, bias, C, M, Nn, K);
    };

    // --- layer 1 -----------------------------------------------------------
    gemm(xb,  W1,   b1,      qkvs1, NN, 2048, 128);     // q|k|v|s
    gemm(eab, We1b, nullptr, e1,    EE, 512,  32);      // edge features

    ZERO(amax1, NN * 4);
    ZERO(den1,  NN * 4);
    ZERO(out1,  NN * 512);

    {
        long long waves = (long long)EE * 4;
        attn_alpha_kernel<<<cdiv(waves * 32, 256), 256, 0, stream>>>(
            qkvs1, 2048, 512, 1024, e1, ei, EE, 4, HID, 0.08838834764831845f,
            alpha1, amax1);
        attn_exp_kernel<<<cdiv(EE * 4, 256), 256, 0, stream>>>(
            alpha1, ei, EE, 4, amax1, den1);
        attn_scatter_kernel<<<cdiv(waves * 32, 256), 256, 0, stream>>>(
            qkvs1, 2048, 1024, e1, ei, EE, 4, HID, alpha1, den1, out1);
    }
    finalize1_kernel<<<cdiv((long long)NN * 512, 256), 256, 0, stream>>>(
        out1, qkvs1, hb, NN * 512);

    // --- layer 2 -----------------------------------------------------------
    gemm(hb,  W2,   b2,      qkvs2, NN, 256, 512);
    gemm(eab, We2b, nullptr, e2,    EE, 64,  32);

    ZERO(amax2, NN);
    ZERO(den2,  NN);
    ZERO(out2,  NN * 64);

    {
        long long waves = (long long)EE;
        attn_alpha_kernel<<<cdiv(waves * 32, 256), 256, 0, stream>>>(
            qkvs2, 256, 64, 128, e2, ei, EE, 1, OUTD, 0.125f, alpha2, amax2);
        attn_exp_kernel<<<cdiv(EE, 256), 256, 0, stream>>>(
            alpha2, ei, EE, 1, amax2, den2);
        attn_scatter_kernel<<<cdiv(waves * 32, 256), 256, 0, stream>>>(
            qkvs2, 256, 128, e2, ei, EE, 1, OUTD, alpha2, den2, out2);
    }
    finalize2_kernel<<<cdiv((long long)NN * 64, 256), 256, 0, stream>>>(
        out2, qkvs2, (float*)d_out, NN * 64);

#undef CVT
#undef CPY
#undef ZERO
}
